// Seq2Seq_47382079209477
// MI455X (gfx1250) — matrix operand order repaired and checked by reading, not compiled, not run on hardware
//
#include <hip/hip_runtime.h>
#include <math.h>

// ---------------------------------------------------------------------------
// Seq2Seq LSTM (encoder/decoder, teacher-forced) for gfx1250.
// All GEMMs run on v_wmma_f32_16x16x32_bf16 with fragment-swizzled operands.
// Big GEMMs use 2Mx4N register blocking (8 WMMA per 12 b128 loads).
// ---------------------------------------------------------------------------

typedef __attribute__((ext_vector_type(16))) __bf16   v16bf;
typedef __attribute__((ext_vector_type(8)))  float    v8f;
typedef __attribute__((ext_vector_type(4)))  unsigned uint4v;

namespace {
constexpr int kS = 64;     // src seq len
constexpr int kT = 48;     // tgt seq len
constexpr int kB = 32;     // batch
constexpr int kE = 512;    // embedding dim
constexpr int kH = 1024;   // hidden
constexpr int kG = 4096;   // 4*H (gates)
constexpr int kO = 32000;  // output dim
constexpr int kDecSteps = kT - 1;          // 47
constexpr int kEncRows  = kS * kB;         // 2048
constexpr int kDecRows  = kDecSteps * kB;  // 1504
}

// float -> bf16 bits, round-to-nearest-even
__device__ __forceinline__ unsigned short f2bf(float f) {
  union { float f; unsigned u; } x; x.f = f;
  unsigned r = x.u + 0x7FFFu + ((x.u >> 16) & 1u);
  return (unsigned short)(r >> 16);
}

__device__ __forceinline__ float sigf(float x) {
  return 1.0f / (1.0f + __expf(-x));
}

// u16 index into A-fragment-swizzled buffer (ISA 7.12.2, 16-bit A 16x32).
__device__ __forceinline__ size_t a_swz(int row, int k, int Kc) {
  int mt = row >> 4, lr = row & 15;
  int kc = k >> 5,  kl = k & 31;
  int half = (kl >> 3) & 1;
  int j    = ((kl >> 4) << 2) + ((kl & 7) >> 1);
  int lane = (half << 4) + lr;
  return ((((size_t)mt * Kc + kc) * 32 + lane) * 16) + (size_t)(j << 1) + (kl & 1);
}

// u16 index into B-fragment-swizzled buffer (16-bit B 32x16).
__device__ __forceinline__ size_t b_swz(int n, int k, int Nt) {
  int nt = n >> 4, nl = n & 15;
  int kc = k >> 5, kl = k & 31;
  int lane = (((kl >> 3) & 1) << 4) + nl;
  int j    = ((kl >> 4) << 2) + ((kl & 7) >> 1);
  return ((((size_t)kc * Nt + nt) * 32 + lane) * 16) + (size_t)(j << 1) + (kl & 1);
}

// ---- weight pack: W[N,K] f32 (used as B = W^T) -> swizzled bf16 fragments
__global__ void pack_b_kernel(const float* __restrict__ W,
                              unsigned short* __restrict__ Bp, int N, int K) {
  long long idx = (long long)blockIdx.x * blockDim.x + threadIdx.x;
  if (idx >= (long long)N * K) return;
  int n = (int)(idx / K);
  int k = (int)(idx - (long long)n * K);
  Bp[b_swz(n, k, N >> 4)] = f2bf(W[idx]);
}

__global__ void bias_sum_kernel(const float* __restrict__ b0,
                                const float* __restrict__ b1,
                                float* __restrict__ bs, int n) {
  int i = blockIdx.x * blockDim.x + threadIdx.x;
  if (i < n) bs[i] = b0[i] + b1[i];
}

// ---- embedding gather -> A-fragment-swizzled bf16
// row r = step*32 + b ; token = tokens[b*tokld + step]
__global__ void embed_swz_kernel(const int* __restrict__ tokens,
                                 const float* __restrict__ emb,
                                 unsigned short* __restrict__ Ap,
                                 int nsteps, int tokld) {
  int idx = blockIdx.x * blockDim.x + threadIdx.x;
  if (idx >= nsteps * kB * kE) return;
  int r = idx >> 9;         // / kE
  int k = idx & (kE - 1);
  int step = r >> 5;
  int b    = r & 31;
  int tok = tokens[b * tokld + step];
  Ap[a_swz(r, k, kE / 32)] = f2bf(emb[(size_t)tok * kE + k]);
}

__global__ void init_state_kernel(float* __restrict__ c,
                                  unsigned short* __restrict__ h) {
  int i = blockIdx.x * blockDim.x + threadIdx.x;
  if (i < kB * kH) { c[i] = 0.0f; h[i] = 0; }
}

// ---- single-tile WMMA GEMM (used for the small sequential h@Whh^T GEMM:
//      Mt=2, Nt=256 -> 512 waves; parallelism over traffic)
__global__ __launch_bounds__(256)
void gemm_bf16_kernel(const unsigned* __restrict__ Ap,
                      const unsigned* __restrict__ Bp,
                      float* __restrict__ C,
                      const float* __restrict__ bias,
                      int Mt, int Nt, int Kc) {
  int gid  = blockIdx.x * blockDim.x + threadIdx.x;
  int wave = gid >> 5;
  int lane = gid & 31;
  if (wave >= Mt * Nt) return;
  int mt = wave / Nt;
  int nt = wave - mt * Nt;

  const unsigned* a = Ap + ((size_t)mt * Kc * 32 + lane) * 8;
  const unsigned* b = Bp + ((size_t)nt * 32 + lane) * 8;
  size_t bstep = (size_t)Nt * 256;

  v8f acc = {};
  for (int kc = 0; kc < Kc; ++kc) {
    union { v16bf v; uint4v q[2]; } fa, fb;
    fa.q[0] = *(const uint4v*)a;
    fa.q[1] = *(const uint4v*)(a + 4);
    fb.q[0] = *(const uint4v*)b;
    fb.q[1] = *(const uint4v*)(b + 4);
    acc = __builtin_amdgcn_wmma_f32_16x16x32_bf16(
        false, fa.v, false, fb.v, (short)0, acc, false, false);
    a += 256;
    b += bstep;
  }

  union { v8f v; float f[8]; } r; r.v = acc;
  int n     = nt * 16 + (lane & 15);
  int mbase = mt * 16 + ((lane >> 4) << 3);
  int ldc   = Nt << 4;
  float bv = bias ? bias[n] : 0.0f;
#pragma unroll
  for (int i = 0; i < 8; ++i)
    C[(size_t)(mbase + i) * ldc + n] = r.f[i] + bv;
}

// ---- 2Mx4N register-blocked WMMA GEMM, plain row-major f32 C (+bias).
//      Requires Mt even, Nt % 4 == 0. 8 WMMA per 12 b128 loads.
__global__ __launch_bounds__(256)
void gemm_blk_kernel(const unsigned* __restrict__ Ap,
                     const unsigned* __restrict__ Bp,
                     float* __restrict__ C,
                     const float* __restrict__ bias,
                     int Mt, int Nt, int Kc) {
  int gid  = blockIdx.x * blockDim.x + threadIdx.x;
  int wave = gid >> 5;
  int lane = gid & 31;
  int MtB = Mt >> 1, NtB = Nt >> 2;
  if (wave >= MtB * NtB) return;
  int mtb = wave / NtB;
  int ntb = wave - mtb * NtB;
  int mt0 = mtb << 1;
  int nt0 = ntb << 2;

  const unsigned* a0 = Ap + ((size_t)mt0 * Kc * 32 + lane) * 8;
  const unsigned* a1 = a0 + (size_t)Kc * 256;   // next M tile
  const unsigned* b  = Bp + ((size_t)nt0 * 32 + lane) * 8;
  size_t bstep = (size_t)Nt * 256;

  v8f acc[2][4] = {};
#pragma unroll 2
  for (int kc = 0; kc < Kc; ++kc) {
    union { v16bf v; uint4v q[2]; } fa[2], fb[4];
    fa[0].q[0] = *(const uint4v*)a0;
    fa[0].q[1] = *(const uint4v*)(a0 + 4);
    fa[1].q[0] = *(const uint4v*)a1;
    fa[1].q[1] = *(const uint4v*)(a1 + 4);
#pragma unroll
    for (int n = 0; n < 4; ++n) {
      fb[n].q[0] = *(const uint4v*)(b + n * 256);
      fb[n].q[1] = *(const uint4v*)(b + n * 256 + 4);
    }
#pragma unroll
    for (int m = 0; m < 2; ++m)
#pragma unroll
      for (int n = 0; n < 4; ++n)
        acc[m][n] = __builtin_amdgcn_wmma_f32_16x16x32_bf16(
            false, fa[m].v, false, fb[n].v, (short)0, acc[m][n], false, false);
    a0 += 256;
    a1 += 256;
    b  += bstep;
  }

  int ldc = Nt << 4;
#pragma unroll
  for (int m = 0; m < 2; ++m) {
#pragma unroll
    for (int n = 0; n < 4; ++n) {
      union { v8f v; float f[8]; } r; r.v = acc[m][n];
      int nn    = (nt0 + n) * 16 + (lane & 15);
      int mbase = (mt0 + m) * 16 + ((lane >> 4) << 3);
      float bv = bias ? bias[nn] : 0.0f;
#pragma unroll
      for (int i = 0; i < 8; ++i)
        C[(size_t)(mbase + i) * ldc + nn] = r.f[i] + bv;
    }
  }
}

// ---- 2Mx4N blocked FC GEMM: scatters into preds[B,T,O] at t+1
__global__ __launch_bounds__(256)
void gemm_blk_fc_kernel(const unsigned* __restrict__ Ap,
                        const unsigned* __restrict__ Bp,
                        float* __restrict__ out,
                        const float* __restrict__ bias,
                        int Mt, int Nt, int Kc) {
  int gid  = blockIdx.x * blockDim.x + threadIdx.x;
  int wave = gid >> 5;
  int lane = gid & 31;
  int MtB = Mt >> 1, NtB = Nt >> 2;
  if (wave >= MtB * NtB) return;
  int mtb = wave / NtB;
  int ntb = wave - mtb * NtB;
  int mt0 = mtb << 1;
  int nt0 = ntb << 2;

  const unsigned* a0 = Ap + ((size_t)mt0 * Kc * 32 + lane) * 8;
  const unsigned* a1 = a0 + (size_t)Kc * 256;
  const unsigned* b  = Bp + ((size_t)nt0 * 32 + lane) * 8;
  size_t bstep = (size_t)Nt * 256;

  v8f acc[2][4] = {};
#pragma unroll 2
  for (int kc = 0; kc < Kc; ++kc) {
    union { v16bf v; uint4v q[2]; } fa[2], fb[4];
    fa[0].q[0] = *(const uint4v*)a0;
    fa[0].q[1] = *(const uint4v*)(a0 + 4);
    fa[1].q[0] = *(const uint4v*)a1;
    fa[1].q[1] = *(const uint4v*)(a1 + 4);
#pragma unroll
    for (int n = 0; n < 4; ++n) {
      fb[n].q[0] = *(const uint4v*)(b + n * 256);
      fb[n].q[1] = *(const uint4v*)(b + n * 256 + 4);
    }
#pragma unroll
    for (int m = 0; m < 2; ++m)
#pragma unroll
      for (int n = 0; n < 4; ++n)
        acc[m][n] = __builtin_amdgcn_wmma_f32_16x16x32_bf16(
            false, fa[m].v, false, fb[n].v, (short)0, acc[m][n], false, false);
    a0 += 256;
    a1 += 256;
    b  += bstep;
  }

#pragma unroll
  for (int m = 0; m < 2; ++m) {
#pragma unroll
    for (int n = 0; n < 4; ++n) {
      union { v8f v; float f[8]; } r; r.v = acc[m][n];
      int nn    = (nt0 + n) * 16 + (lane & 15);
      int mbase = (mt0 + m) * 16 + ((lane >> 4) << 3);
      float bv = bias[nn];
#pragma unroll
      for (int i = 0; i < 8; ++i) {
        int mg = mbase + i;            // global row = t*32 + b
        int t  = mg >> 5;
        int bb = mg & 31;
        out[(size_t)bb * (kT * kO) + (size_t)(t + 1) * kO + nn] = r.f[i] + bv;
      }
    }
  }
}

// ---- fused LSTM gate update: c,h from Xg (x@Wih^T + biases) + Ghh (h@Whh^T)
__global__ void gates_kernel(const float* __restrict__ Xg,
                             const float* __restrict__ Ghh,
                             float* __restrict__ c,
                             unsigned short* __restrict__ h_swz,
                             unsigned short* __restrict__ Hd_swz, int t) {
  int idx = blockIdx.x * blockDim.x + threadIdx.x;
  if (idx >= kB * kH) return;
  int b = idx >> 10;
  int j = idx & (kH - 1);
  size_t base = (size_t)b * kG + j;
  float gi = Xg[base]            + Ghh[base];
  float gf = Xg[base + kH]       + Ghh[base + kH];
  float gg = Xg[base + 2 * kH]   + Ghh[base + 2 * kH];
  float go = Xg[base + 3 * kH]   + Ghh[base + 3 * kH];
  float cn = sigf(gf) * c[idx] + sigf(gi) * tanhf(gg);
  float hn = sigf(go) * tanhf(cn);
  c[idx] = cn;
  unsigned short hb = f2bf(hn);
  h_swz[a_swz(b, j, kH / 32)] = hb;
  if (Hd_swz) Hd_swz[a_swz(t * kB + b, j, kH / 32)] = hb;
}

// ---- preds[:, 0, :] = 0
__global__ void zero_t0_kernel(float* __restrict__ out) {
  int idx = blockIdx.x * blockDim.x + threadIdx.x;
  if (idx >= kB * kO) return;
  int b = idx / kO;
  int n = idx - b * kO;
  out[(size_t)b * (kT * kO) + n] = 0.0f;
}

static inline int nblk(long long n) { return (int)((n + 255) / 256); }

extern "C" void kernel_launch(void* const* d_in, const int* in_sizes, int n_in,
                              void* d_out, int out_size, void* d_ws, size_t ws_size,
                              hipStream_t stream) {
  (void)in_sizes; (void)n_in; (void)out_size; (void)ws_size;
  const int*   x       = (const int*)d_in[0];
  const int*   label   = (const int*)d_in[1];
  // d_in[2] = teaching_p == 1 -> always teacher-forced; argmax path is dead.
  const float* enc_emb = (const float*)d_in[3];
  const float* enc_Wih = (const float*)d_in[4];
  const float* enc_Whh = (const float*)d_in[5];
  const float* enc_bih = (const float*)d_in[6];
  const float* enc_bhh = (const float*)d_in[7];
  const float* dec_emb = (const float*)d_in[8];
  const float* dec_Wih = (const float*)d_in[9];
  const float* dec_Whh = (const float*)d_in[10];
  const float* dec_bih = (const float*)d_in[11];
  const float* dec_bhh = (const float*)d_in[12];
  const float* fc_W    = (const float*)d_in[13];
  const float* fc_b    = (const float*)d_in[14];
  float* out = (float*)d_out;

  char* ws = (char*)d_ws;
  size_t off = 0;
  auto take = [&](size_t bytes) -> char* {
    char* p = ws + off;
    off = (off + bytes + 255) & ~(size_t)255;
    return p;
  };

  unsigned short* encWihP = (unsigned short*)take((size_t)kG * kE * 2);
  unsigned short* encWhhP = (unsigned short*)take((size_t)kG * kH * 2);
  unsigned short* decWihP = (unsigned short*)take((size_t)kG * kE * 2);
  unsigned short* decWhhP = (unsigned short*)take((size_t)kG * kH * 2);
  unsigned short* fcWP    = (unsigned short*)take((size_t)kO * kH * 2);
  unsigned short* xeEnc   = (unsigned short*)take((size_t)kEncRows * kE * 2);
  unsigned short* xeDec   = (unsigned short*)take((size_t)kDecRows * kE * 2);
  float*          xgBuf   = (float*)take((size_t)kEncRows * kG * 4);  // reused enc->dec
  float*          ghh     = (float*)take((size_t)kB * kG * 4);
  unsigned short* hSwz    = (unsigned short*)take((size_t)kB * kH * 2);
  float*          cState  = (float*)take((size_t)kB * kH * 4);
  unsigned short* hdSwz   = (unsigned short*)take((size_t)kDecRows * kH * 2);
  float*          biasE   = (float*)take((size_t)kG * 4);
  float*          biasD   = (float*)take((size_t)kG * 4);

  // ---- one-time packs (weights -> swizzled bf16 B fragments)
  pack_b_kernel<<<nblk((long long)kG * kE), 256, 0, stream>>>(enc_Wih, encWihP, kG, kE);
  pack_b_kernel<<<nblk((long long)kG * kH), 256, 0, stream>>>(enc_Whh, encWhhP, kG, kH);
  pack_b_kernel<<<nblk((long long)kG * kE), 256, 0, stream>>>(dec_Wih, decWihP, kG, kE);
  pack_b_kernel<<<nblk((long long)kG * kH), 256, 0, stream>>>(dec_Whh, decWhhP, kG, kH);
  pack_b_kernel<<<nblk((long long)kO * kH), 256, 0, stream>>>(fc_W,    fcWP,    kO, kH);
  bias_sum_kernel<<<nblk(kG), 256, 0, stream>>>(enc_bih, enc_bhh, biasE, kG);
  bias_sum_kernel<<<nblk(kG), 256, 0, stream>>>(dec_bih, dec_bhh, biasD, kG);

  // ---- embedding gathers (swizzled A fragments)
  embed_swz_kernel<<<nblk((long long)kEncRows * kE), 256, 0, stream>>>(
      x, enc_emb, xeEnc, kS, kS);
  embed_swz_kernel<<<nblk((long long)kDecRows * kE), 256, 0, stream>>>(
      label, dec_emb, xeDec, kDecSteps, kT);

  init_state_kernel<<<nblk(kB * kH), 256, 0, stream>>>(cState, hSwz);

  // ---- batched input-projection GEMM, encoder: [2048,512]x[512,4096]+bias
  {
    int Mt = kEncRows / 16, Nt = kG / 16, Kc = kE / 32;
    long long waves = (long long)(Mt / 2) * (Nt / 4);
    gemm_blk_kernel<<<nblk(waves * 32), 256, 0, stream>>>(
        (const unsigned*)xeEnc, (const unsigned*)encWihP, xgBuf, biasE, Mt, Nt, Kc);
  }

  // ---- encoder recurrence: 64 sequential (GEMM -> gates) steps
  {
    int Mt = kB / 16, Nt = kG / 16, Kc = kH / 32;
    long long waves = (long long)Mt * Nt;  // 512
    for (int s = 0; s < kS; ++s) {
      gemm_bf16_kernel<<<nblk(waves * 32), 256, 0, stream>>>(
          (const unsigned*)hSwz, (const unsigned*)encWhhP, ghh, nullptr, Mt, Nt, Kc);
      gates_kernel<<<nblk(kB * kH), 256, 0, stream>>>(
          xgBuf + (size_t)s * kB * kG, ghh, cState, hSwz, nullptr, 0);
    }
  }

  // ---- batched input-projection GEMM, decoder (reuses xgBuf)
  {
    int Mt = kDecRows / 16, Nt = kG / 16, Kc = kE / 32;
    long long waves = (long long)(Mt / 2) * (Nt / 4);
    gemm_blk_kernel<<<nblk(waves * 32), 256, 0, stream>>>(
        (const unsigned*)xeDec, (const unsigned*)decWihP, xgBuf, biasD, Mt, Nt, Kc);
  }

  // ---- decoder recurrence: 47 sequential steps; h_t archived into hdSwz
  {
    int Mt = kB / 16, Nt = kG / 16, Kc = kH / 32;
    long long waves = (long long)Mt * Nt;
    for (int t = 0; t < kDecSteps; ++t) {
      gemm_bf16_kernel<<<nblk(waves * 32), 256, 0, stream>>>(
          (const unsigned*)hSwz, (const unsigned*)decWhhP, ghh, nullptr, Mt, Nt, Kc);
      gates_kernel<<<nblk(kB * kH), 256, 0, stream>>>(
          xgBuf + (size_t)t * kB * kG, ghh, cState, hSwz, hdSwz, t);
    }
  }

  // ---- one big FC GEMM: [1504,1024]x[1024,32000] -> preds[:,1:,:]
  {
    int Mt = kDecRows / 16, Nt = kO / 16, Kc = kH / 32;
    long long waves = (long long)(Mt / 2) * (Nt / 4);  // 23500
    gemm_blk_fc_kernel<<<nblk(waves * 32), 256, 0, stream>>>(
        (const unsigned*)hdSwz, (const unsigned*)fcWP, out, fc_b, Mt, Nt, Kc);
  }

  // ---- preds[:, 0, :] = 0
  zero_t0_kernel<<<nblk((long long)kB * kO), 256, 0, stream>>>(out);
}
